// SpGAT_22909355557429
// MI455X (gfx1250) — matrix operand order repaired and checked by reading, not compiled, not run on hardware
//
#include <hip/hip_runtime.h>

// ---------------------------------------------------------------------------
// SpGAT on MI455X (gfx1250, wave32).
//   Phase 1: h = bf16(x) @ bf16(W_all)          -- v_wmma_f32_16x16x32_bf16
//   Phase 2: per-node s/d dots, per-edge exp(-leakyrelu), f32 atomic scatter
//   Phase 3: h' = elu(num/rowsum) -> bf16 ; out layer GEMM + second edge pass
// GEMMs are compute-trivial at 23.3 TB/s HBM; the edge scatter (~819M f32
// atomics, ~13 GB gather traffic) dominates, so h is kept bf16 (102 MB, ~L2
// resident) and atomics are non-returning agent-scope global_atomic_add_f32.
// GEMM inner loop is register double-buffered so B/A-frag loads for K-step
// kt+1 overlap the 8 back-to-back WMMAs of K-step kt.
// ---------------------------------------------------------------------------

typedef unsigned int        uint_t;
typedef unsigned short      u16;
typedef __attribute__((ext_vector_type(16))) __bf16  v16bf;
typedef __attribute__((ext_vector_type(8)))  float   v8f;
typedef __attribute__((ext_vector_type(8)))  uint_t  v8u;
typedef __attribute__((ext_vector_type(4)))  uint_t  v4u;

#define N_NODES 100000
#define N_EDGES 1600000
#define NFEAT   256
#define NHID    64
#define NHEADS  8
#define HTOT    512            // NHEADS * NHID
#define NCLASS  40
#define NCLASSP 64             // padded output cols for GEMM2

__device__ __forceinline__ float bf2f(u16 u) {
    uint_t x = ((uint_t)u) << 16;
    return __builtin_bit_cast(float, x);
}
__device__ __forceinline__ u16 f2bf(float f) {
    uint_t b = __builtin_bit_cast(uint_t, f);
    uint_t r = b + 0x7FFFu + ((b >> 16) & 1u);   // round-to-nearest-even
    return (u16)(r >> 16);
}
__device__ __forceinline__ void atomicAddF32(float* p, float v) {
    (void)__hip_atomic_fetch_add(p, v, __ATOMIC_RELAXED, __HIP_MEMORY_SCOPE_AGENT);
}
__device__ __forceinline__ float eluf(float v) {
    return v > 0.0f ? v : (__expf(v) - 1.0f);
}

// -------------------------- conversion / packing ---------------------------

__global__ void cvt_x_kernel(const float* __restrict__ x, u16* __restrict__ xb, int n) {
    int i = blockIdx.x * blockDim.x + threadIdx.x;
    if (i < n) xb[i] = f2bf(x[i]);
}

// Pack W_heads [8,256,64] f32 -> fragment-ordered bf16 B for GEMM1.
// BP index = ((nt*KSTEPS + kt)*32 + lane)*16 + j ; KSTEPS=8, nt in [0,32)
// B-frag layout (16-bit, 32x16): lanes 0-15 col n hold K=kt*32+{0..7,16..23},
// lanes 16-31 hold K=kt*32+{8..15,24..31}.
__global__ void build_bp1_kernel(const float* __restrict__ Wh, u16* __restrict__ BP) {
    int idx = blockIdx.x * blockDim.x + threadIdx.x;        // < 32*8*32*16 = 131072
    if (idx >= 32 * 8 * 32 * 16) return;
    int j    = idx & 15;
    int lane = (idx >> 4) & 31;
    int kt   = (idx >> 9) & 7;
    int nt   = idx >> 12;
    int n = nt * 16 + (lane & 15);
    int k = kt * 32 + ((lane >> 4) << 3) + (j & 7) + ((j >> 3) << 4);
    int head = n >> 6, f = n & 63;
    float v = Wh[(size_t)head * NFEAT * NHID + (size_t)k * NHID + f];
    BP[idx] = f2bf(v);
}

// Pack W_out [512,40] f32 -> fragment-ordered bf16, padded to 64 cols. KSTEPS=16.
__global__ void build_bp2_kernel(const float* __restrict__ Wo, u16* __restrict__ BP) {
    int idx = blockIdx.x * blockDim.x + threadIdx.x;        // < 4*16*32*16 = 32768
    if (idx >= 4 * 16 * 32 * 16) return;
    int j    = idx & 15;
    int lane = (idx >> 4) & 31;
    int kt   = (idx >> 9) & 15;
    int nt   = idx >> 13;
    int n = nt * 16 + (lane & 15);
    int k = kt * 32 + ((lane >> 4) << 3) + (j & 7) + ((j >> 3) << 4);
    float v = (n < NCLASS) ? Wo[(size_t)k * NCLASS + n] : 0.0f;
    BP[idx] = f2bf(v);
}

// ------------------------------- WMMA GEMM ---------------------------------
// A-frag per ISA 16-bit A layout: lane row m = lane&15,
// kbase = kt*32 + (lane>=16 ? 8 : 0); elems 0..7 = K kbase..+7,
// elems 8..15 = K kbase+16..+23  -> two contiguous 16B loads.
__device__ __forceinline__ v16bf load_afrag(const u16* __restrict__ ap) {
    v4u a0 = *(const v4u*)(ap);
    v4u a1 = *(const v4u*)(ap + 16);
    v8u au;
    au[0] = a0[0]; au[1] = a0[1]; au[2] = a0[2]; au[3] = a0[3];
    au[4] = a1[0]; au[5] = a1[1]; au[6] = a1[2]; au[7] = a1[3];
    return __builtin_bit_cast(v16bf, au);
}
__device__ __forceinline__ v16bf load_bfrag(const u16* __restrict__ bp) {
    v8u bu = *(const v8u*)bp;
    return __builtin_bit_cast(v16bf, bu);
}

// C[Nrows, Ncols] = A[Nrows, K]bf16 @ B ; workgroup = 8 waves = 128 rows,
// wave tile = 16 x (NFRAG*16). Register double-buffered over K-steps.
template <int KSTEPS, int NFRAG, bool BF16OUT>
__global__ __launch_bounds__(256) void gemm_wmma_kernel(
    const u16* __restrict__ A, const u16* __restrict__ BP,
    void* __restrict__ Cout, int Nrows, int Ncols)
{
    constexpr int K = KSTEPS * 32;
    const int lane = threadIdx.x & 31;
    const int wave = threadIdx.x >> 5;
    const int rowBase = blockIdx.x * 128 + wave * 16;
    const int colBase = blockIdx.y * (NFRAG * 16);
    int m = rowBase + (lane & 15);
    if (m >= Nrows) m = Nrows - 1;                 // clamp; stores are masked
    const int kShift = (lane >> 4) * 8;

    const u16* __restrict__ aBase = A + (size_t)m * K + kShift;
    // wave's B fragments: BP + ((ntg*KSTEPS + kt)*32 + lane)*16
    const u16* __restrict__ bBase =
        BP + ((size_t)(blockIdx.y * NFRAG) * KSTEPS * 32 + lane) * 16;

    v8f acc[NFRAG] = {};

    // preload K-step 0
    v16bf aCur = load_afrag(aBase);
    v16bf bCur[NFRAG];
#pragma unroll
    for (int nf = 0; nf < NFRAG; ++nf)
        bCur[nf] = load_bfrag(bBase + (size_t)nf * KSTEPS * 32 * 16);

#pragma unroll
    for (int kt = 0; kt < KSTEPS - 1; ++kt) {
        // issue next K-step's loads first so they overlap this step's WMMAs
        v16bf aNxt = load_afrag(aBase + (kt + 1) * 32);
        v16bf bNxt[NFRAG];
#pragma unroll
        for (int nf = 0; nf < NFRAG; ++nf)
            bNxt[nf] = load_bfrag(bBase + ((size_t)nf * KSTEPS + (kt + 1)) * 32 * 16);
#pragma unroll
        for (int nf = 0; nf < NFRAG; ++nf)
            acc[nf] = __builtin_amdgcn_wmma_f32_16x16x32_bf16(
                false, aCur, false, bCur[nf], (short)0, acc[nf], false, false);
        aCur = aNxt;
#pragma unroll
        for (int nf = 0; nf < NFRAG; ++nf) bCur[nf] = bNxt[nf];
    }
#pragma unroll
    for (int nf = 0; nf < NFRAG; ++nf)
        acc[nf] = __builtin_amdgcn_wmma_f32_16x16x32_bf16(
            false, aCur, false, bCur[nf], (short)0, acc[nf], false, false);

    // C/D layout: vgpr r, lane L -> row rowBase + r + (L>=16 ? 8:0), col = colBase + (L&15)
    const int rOff = (lane >> 4) * 8;
    const int cl = lane & 15;
#pragma unroll
    for (int nf = 0; nf < NFRAG; ++nf) {
        int col = colBase + nf * 16 + cl;
#pragma unroll
        for (int r = 0; r < 8; ++r) {
            int row = rowBase + rOff + r;
            if (row < Nrows) {
                float v = acc[nf][r];
                if constexpr (BF16OUT)
                    ((u16*)Cout)[(size_t)row * Ncols + col] = f2bf(v);
                else
                    ((float*)Cout)[(size_t)row * Ncols + col] = v;
            }
        }
    }
}

// ----------------------- per-node attention dots ---------------------------

__global__ void sd1_kernel(const u16* __restrict__ hb, const float* __restrict__ a_heads,
                           float* __restrict__ s1, float* __restrict__ d1) {
    int t = blockIdx.x * blockDim.x + threadIdx.x;
    if (t >= N_NODES * NHEADS) return;
    int n = t >> 3, head = t & 7;
    const u16* hp = hb + (size_t)n * HTOT + head * NHID;
    const float* as = a_heads + head * 2 * NHID;
    float s = 0.0f, d = 0.0f;
#pragma unroll
    for (int v8 = 0; v8 < NHID / 8; ++v8) {        // 8 x b128 vector loads
        v4u pk = *(const v4u*)(hp + v8 * 8);
#pragma unroll
        for (int e = 0; e < 4; ++e) {
            float h0 = bf2f((u16)(pk[e] & 0xFFFFu));
            float h1 = bf2f((u16)(pk[e] >> 16));
            int f = v8 * 8 + e * 2;
            s += h0 * as[f]        + h1 * as[f + 1];
            d += h0 * as[NHID + f] + h1 * as[NHID + f + 1];
        }
    }
    s1[head * N_NODES + n] = s;
    d1[head * N_NODES + n] = d;
}

__global__ void sd2_kernel(const float* __restrict__ h2, const float* __restrict__ a_out,
                           float* __restrict__ s2, float* __restrict__ d2) {
    int n = blockIdx.x * blockDim.x + threadIdx.x;
    if (n >= N_NODES) return;
    const float* hp = h2 + (size_t)n * NCLASSP;
    float s = 0.0f, d = 0.0f;
#pragma unroll 8
    for (int c = 0; c < NCLASS; ++c) {
        float h = hp[c];
        s += h * a_out[c];
        d += h * a_out[NCLASS + c];
    }
    s2[n] = s;
    d2[n] = d;
}

// ----------------------------- edge scatter --------------------------------
// 64 threads per (edge, head) unit; lane = feature.
__global__ __launch_bounds__(256) void scatter1_kernel(
    const int* __restrict__ src, const int* __restrict__ dst,
    const float* __restrict__ s1, const float* __restrict__ d1,
    const u16* __restrict__ hb, float* __restrict__ num, float* __restrict__ rowsum)
{
    long long unit = (long long)blockIdx.x * 4 + (threadIdx.x >> 6);
    int f = threadIdx.x & 63;
    int edge = (int)(unit >> 3);
    int head = (int)(unit & 7);
    if (edge >= N_EDGES) return;
    int s = src[edge], d = dst[edge];
    const u16* hp = hb + (size_t)d * HTOT + head * NHID;
    __builtin_prefetch(hp, 0, 0);                 // global_prefetch_b8
    float lg = s1[head * N_NODES + s] + d1[head * N_NODES + d];
    float lr = lg > 0.0f ? lg : 0.2f * lg;        // leaky_relu slope 0.2
    float e = __expf(-lr);
    float hv = bf2f(hp[f]);
    atomicAddF32(&num[(size_t)s * HTOT + head * NHID + f], e * hv);
    if (f == 0) atomicAddF32(&rowsum[head * N_NODES + s], e);
}

__global__ __launch_bounds__(256) void scatter2_kernel(
    const int* __restrict__ src, const int* __restrict__ dst,
    const float* __restrict__ s2, const float* __restrict__ d2,
    const float* __restrict__ h2, float* __restrict__ num2, float* __restrict__ rowsum2)
{
    int edge = blockIdx.x * 4 + (threadIdx.x >> 6);
    int f = threadIdx.x & 63;
    if (edge >= N_EDGES) return;
    int s = src[edge], d = dst[edge];
    const float* hp = h2 + (size_t)d * NCLASSP;
    __builtin_prefetch(hp, 0, 0);
    float lg = s2[s] + d2[d];
    float lr = lg > 0.0f ? lg : 0.2f * lg;
    float e = __expf(-lr);
    if (f < NCLASS) atomicAddF32(&num2[(size_t)s * NCLASS + f], e * hp[f]);
    if (f == 0) atomicAddF32(&rowsum2[s], e);
}

// --------------------------- normalize + elu -------------------------------

__global__ void div_elu1_kernel(const float* __restrict__ num, const float* __restrict__ rowsum,
                                u16* __restrict__ hb) {
    int idx = blockIdx.x * blockDim.x + threadIdx.x;
    if (idx >= N_NODES * HTOT) return;
    int n = idx >> 9;            // / 512
    int c = idx & 511;
    int head = c >> 6;
    float v = num[idx] / (rowsum[head * N_NODES + n] + 1e-16f);
    hb[idx] = f2bf(eluf(v));     // concat=True -> elu
}

__global__ void final_kernel(const float* __restrict__ num2, const float* __restrict__ rowsum2,
                             float* __restrict__ out) {
    int idx = blockIdx.x * blockDim.x + threadIdx.x;
    if (idx >= N_NODES * NCLASS) return;
    int n = idx / NCLASS;
    float v = num2[idx] / (rowsum2[n] + 1e-16f);
    out[idx] = eluf(v);          // final elu
}

// ------------------------------ host driver --------------------------------

static inline size_t alignUp(size_t v, size_t a) { return (v + a - 1) & ~(a - 1); }

extern "C" void kernel_launch(void* const* d_in, const int* in_sizes, int n_in,
                              void* d_out, int out_size, void* d_ws, size_t ws_size,
                              hipStream_t stream) {
    (void)in_sizes; (void)n_in; (void)out_size; (void)ws_size;

    const float* x        = (const float*)d_in[0];   // [N, 256]
    const int*   e_src    = (const int*)d_in[1];     // edge_index[0]
    const int*   e_dst    = ((const int*)d_in[1]) + N_EDGES;
    const float* W_heads  = (const float*)d_in[2];   // [8, 256, 64]
    const float* a_heads  = (const float*)d_in[3];   // [8, 128]
    const float* W_out    = (const float*)d_in[4];   // [512, 40]
    const float* a_out    = (const float*)d_in[5];   // [80]
    float*       out      = (float*)d_out;           // [N, 40]

    char* w = (char*)d_ws;
    size_t off = 0;
    // --- zeroed accumulator region (one memset) ---
    float* num1    = (float*)(w + off); off += alignUp((size_t)N_NODES * HTOT * 4, 256);
    float* rowsum1 = (float*)(w + off); off += alignUp((size_t)NHEADS * N_NODES * 4, 256);
    float* num2    = (float*)(w + off); off += alignUp((size_t)N_NODES * NCLASS * 4, 256);
    float* rowsum2 = (float*)(w + off); off += alignUp((size_t)N_NODES * 4, 256);
    size_t zeroBytes = off;
    // --- non-zeroed scratch ---
    u16*   xb   = (u16*)(w + off); off += alignUp((size_t)N_NODES * NFEAT * 2, 256);
    u16*   BP1  = (u16*)(w + off); off += alignUp((size_t)32 * 8 * 32 * 16 * 2, 256);
    u16*   BP2  = (u16*)(w + off); off += alignUp((size_t)4 * 16 * 32 * 16 * 2, 256);
    u16*   hb   = (u16*)(w + off); off += alignUp((size_t)N_NODES * HTOT * 2, 256);
    float* s1   = (float*)(w + off); off += alignUp((size_t)NHEADS * N_NODES * 4, 256);
    float* d1   = (float*)(w + off); off += alignUp((size_t)NHEADS * N_NODES * 4, 256);
    float* h2   = (float*)(w + off); off += alignUp((size_t)N_NODES * NCLASSP * 4, 256);
    float* s2   = (float*)(w + off); off += alignUp((size_t)N_NODES * 4, 256);
    float* d2   = (float*)(w + off); off += alignUp((size_t)N_NODES * 4, 256);

    hipMemsetAsync(w, 0, zeroBytes, stream);

    // x -> bf16
    {
        int n = N_NODES * NFEAT;
        cvt_x_kernel<<<(n + 255) / 256, 256, 0, stream>>>(x, xb, n);
    }
    // pack B matrices into WMMA fragment order
    build_bp1_kernel<<<(32 * 8 * 32 * 16) / 256, 256, 0, stream>>>(W_heads, BP1);
    build_bp2_kernel<<<(4 * 16 * 32 * 16) / 256, 256, 0, stream>>>(W_out, BP2);

    // GEMM1: hb[N,512]bf16 = xb[N,256] @ W_all[256,512]
    {
        dim3 grid((N_NODES + 127) / 128, HTOT / 128);
        gemm_wmma_kernel<8, 8, true><<<grid, 256, 0, stream>>>(xb, BP1, hb, N_NODES, HTOT);
    }
    // per-node attention dot products, layer 1
    sd1_kernel<<<(N_NODES * NHEADS + 255) / 256, 256, 0, stream>>>(hb, a_heads, s1, d1);

    // edge scatter, layer 1: E*8 units of 64 threads
    {
        long long units = (long long)N_EDGES * NHEADS;
        int blocks = (int)((units + 3) / 4);
        scatter1_kernel<<<blocks, 256, 0, stream>>>(e_src, e_dst, s1, d1, hb, num1, rowsum1);
    }
    // h' = elu(num/rowsum) -> bf16 (overwrites hb)
    div_elu1_kernel<<<(N_NODES * HTOT + 255) / 256, 256, 0, stream>>>(num1, rowsum1, hb);

    // GEMM2: h2[N,64]f32 = hb[N,512] @ W_out[512,40 pad 64]
    {
        dim3 grid((N_NODES + 127) / 128, 1);
        gemm_wmma_kernel<16, 4, false><<<grid, 256, 0, stream>>>(hb, BP2, h2, N_NODES, NCLASSP);
    }
    sd2_kernel<<<(N_NODES + 255) / 256, 256, 0, stream>>>(h2, a_out, s2, d2);

    // edge scatter, layer 2: E units of 64 threads (40 active)
    scatter2_kernel<<<(N_EDGES + 3) / 4, 256, 0, stream>>>(e_src, e_dst, s2, d2, h2, num2, rowsum2);

    // out = elu(num2/rowsum2)
    final_kernel<<<(N_NODES * NCLASS + 255) / 256, 256, 0, stream>>>(num2, rowsum2, out);
}
